// BidirectionalSeqLEM_54820962567003
// MI455X (gfx1250) — compile-verified, matrix operand order checked by reading, not run on hardware
//
#include <hip/hip_runtime.h>

// ---------------------------------------------------------------------------
// Bidirectional LEM scan for MI455X (gfx1250).
// Latency-bound recurrence -> persistent kernel: 8 workgroups
// (2 directions x 4 batch-groups of 16 rows; batch rows never mix so there is
// no inter-workgroup communication). 512 threads = 16 waves per WG.
// Recurrent state in registers/LDS; weights pre-swizzled to bf16 WMMA
// B-fragment layout in workspace and re-streamed from L2 each step (an asm
// clobber defeats LICM so they are NOT hoisted+spilled to scratch).
// Input projection for step t+1 is software-pipelined behind the recurrent
// matmuls; x slabs are prefetched into LDS with global_load_async_to_lds_b128
// (ASYNCcnt) one step ahead.
// ---------------------------------------------------------------------------

typedef __attribute__((ext_vector_type(16))) __bf16 bf16x16;
typedef __attribute__((ext_vector_type(8)))  __bf16 bf16x8;
typedef __attribute__((ext_vector_type(2)))  __bf16 bf16x2;
typedef __attribute__((ext_vector_type(8)))  float  f32x8;
typedef __attribute__((ext_vector_type(4)))  float  f32x4;

#define TT    2048
#define BATCH 64
#define KCN   8            // 256/32 K-chunks per matmul

// Per-direction swizzled-weight layout in workspace (bf16 elements).
#define WIH_ELEMS 262144   // 64 n-tiles * 8 kc * 512
#define WHH_ELEMS 196608   // 48 n-tiles * 8 kc * 512
#define DIR_STRIDE 524288  // + WzF 65536 -> 1 MB per direction as bf16

// Dynamic-LDS carve (bytes)
#define TI_OFF   0              // bf16 ti[2][1024][16] column-major  (64 KB)
#define Y_OFF    65536          // bf16 y[16][256]                    ( 8 KB)
#define Z_OFF    73728          // bf16 z[16][256]                    ( 8 KB)
#define XB_OFF   81920          // f32  x slabs [2][16][256]          (32 KB)
#define BIAS_OFF 114688         // f32  bih|bhh|bz (2048)             ( 8 KB)
#define SMEM_BYTES 122880

// A/B 16x32 bf16 fragment: lane half g owns K = {8g..8g+7} U {16+8g..16+8g+7}
// (two contiguous 8-element runs -> b128-friendly).

__device__ __forceinline__ bf16x16 load_a_lds(const __bf16* base, int lane,
                                              int k0) {
    const int g = lane >> 4;
    const __bf16* p = base + (lane & 15) * 256 + k0 + 8 * g;
    bf16x8 lo = *(const bf16x8*)(p);
    bf16x8 hi = *(const bf16x8*)(p + 16);
    return __builtin_shufflevector(lo, hi, 0, 1, 2, 3, 4, 5, 6, 7,
                                   8, 9, 10, 11, 12, 13, 14, 15);
}

__device__ __forceinline__ bf16x16 load_a_f32(const float* p0, int lane,
                                              int k0) {
    const int g = lane >> 4;
    const float* p = p0 + (lane & 15) * 256 + k0 + 8 * g;
    f32x4 q0 = *(const f32x4*)(p);
    f32x4 q1 = *(const f32x4*)(p + 4);
    f32x4 q2 = *(const f32x4*)(p + 16);
    f32x4 q3 = *(const f32x4*)(p + 20);
    bf16x16 a;
#pragma unroll
    for (int i = 0; i < 4; ++i) {
        a[i]      = (__bf16)q0[i];
        a[4 + i]  = (__bf16)q1[i];
        a[8 + i]  = (__bf16)q2[i];
        a[12 + i] = (__bf16)q3[i];
    }
    return a;
}

__device__ __forceinline__ bf16x16 load_frag(const __bf16* fbase, int lane) {
    return *(const bf16x16*)(fbase + lane * 16);
}

__device__ __forceinline__ float sigm(float v) {
    return 1.0f / (1.0f + __expf(-v));
}

// Async copy of one x slab (16 rows x 256 f32 = 16 KB) into LDS.
// 512 threads x 2 x b128, tracked by ASYNCcnt.
__device__ __forceinline__ void async_load_x(const float* gsrc, float* ldst,
                                             int tid) {
    unsigned lbase = (unsigned)(size_t)(void*)ldst;   // low 32 bits = LDS addr
    unsigned long long gbase = (unsigned long long)(size_t)gsrc;
#pragma unroll
    for (int i = 0; i < 2; ++i) {
        unsigned loff = lbase + (unsigned)((tid + i * 512) * 16);
        unsigned long long g = gbase + (unsigned long long)((tid + i * 512) * 16);
        asm volatile("global_load_async_to_lds_b128 %0, %1, off"
                     :: "v"(loff), "v"(g) : "memory");
    }
}

__device__ __forceinline__ void wait_async() {
    asm volatile("s_wait_asynccnt 0" ::: "memory");
}

// ---------------------------------------------------------------------------
// Weight pre-swizzle: fp32 [rows x 256] -> bf16 fragment layout
// [(nt*8+kc)][lane][16].  One block per fragment, 2 elems/thread.
// ---------------------------------------------------------------------------
__global__ void swizzle_w(const float* __restrict__ src,
                          __bf16* __restrict__ dst) {
    const int fidx = blockIdx.x;        // nt*8 + kc
    const int nt = fidx >> 3, kc = fidx & 7;
#pragma unroll
    for (int s = 0; s < 2; ++s) {
        int e512 = threadIdx.x * 2 + s;
        int l = e512 >> 4, e = e512 & 15;
        int g = l >> 4;
        int K = (e < 8) ? (8 * g + e) : (16 + 8 * g + (e - 8));
        dst[(size_t)fidx * 512 + e512] =
            (__bf16)src[(nt * 16 + (l & 15)) * 256 + kc * 32 + K];
    }
}

// ---------------------------------------------------------------------------
// Persistent scan: grid = 8 (dir = blockIdx&1, batch group = blockIdx>>1),
// 512 threads = 16 waves. Per wave/step: 3 Whh + 1 Wz + 4 proj tiles,
// K=256 each -> 64 WMMAs.
// ---------------------------------------------------------------------------
__global__ __launch_bounds__(512)
void lem_scan(const float* __restrict__ x,
              const __bf16* __restrict__ wf,
              const float* __restrict__ fbih, const float* __restrict__ fbhh,
              const float* __restrict__ fbz,
              const float* __restrict__ bbih, const float* __restrict__ bbhh,
              const float* __restrict__ bbz,
              float* __restrict__ out) {
    extern __shared__ char smem[];
    __bf16* ti    = (__bf16*)(smem + TI_OFF);   // [2][col*16 + b] col-major
    __bf16* y_lds = (__bf16*)(smem + Y_OFF);    // [b][256]
    __bf16* z_lds = (__bf16*)(smem + Z_OFF);    // [b][256]
    float*  xb    = (float*)(smem + XB_OFF);    // [2][16][256]
    float*  bias  = (float*)(smem + BIAS_OFF);  // bih[1024]|bhh[768]|bz[256]

    const int dir  = blockIdx.x & 1;
    const int bg   = blockIdx.x >> 1;
    const int tid  = threadIdx.x;
    const int lane = tid & 31;
    const int w    = tid >> 5;              // wave 0..15
    const int g    = lane >> 4;
    const int ln   = lane & 15;

    // Mutable weight-fragment bases (re-derived each step via asm clobber).
    const __bf16* WihF = wf + (size_t)dir * DIR_STRIDE;
    const __bf16* WhhF = WihF + WIH_ELEMS;
    const __bf16* WzF  = WhhF + WHH_ELEMS;
    const float* bihp = dir ? bbih : fbih;
    const float* bhhp = dir ? bbhh : fbhh;
    const float* bzp  = dir ? bbz  : fbz;

    for (int i = tid; i < 1024; i += 512) bias[i] = bihp[i];
    for (int i = tid; i < 768;  i += 512) bias[1024 + i] = bhhp[i];
    for (int i = tid; i < 256;  i += 512) bias[1792 + i] = bzp[i];
    for (int i = tid; i < 4096; i += 512) y_lds[i] = (__bf16)0.0f;

    // Lane owns (b = r + 8g, j = w*16 + ln), r = 0..7.
    float zreg[8], yreg[8], msb[8];
#pragma unroll
    for (int i = 0; i < 8; ++i) { zreg[i] = 0.f; yreg[i] = 0.f; msb[i] = 0.f; }
    __syncthreads();

    // ---- Prologue: ti(0) from global x, async-stage x(1) into LDS ----
    {
        const int ts0 = dir ? (TT - 1) : 0;
        const float* xt0 = x + (size_t)ts0 * (BATCH * 256) + bg * (16 * 256);
        bf16x16 ax[KCN];
#pragma unroll
        for (int kc = 0; kc < KCN; ++kc) ax[kc] = load_a_f32(xt0, lane, kc * 32);
#pragma unroll
        for (int i = 0; i < 4; ++i) {
            const int nt = w * 4 + i;
            f32x8 c = {};
#pragma unroll
            for (int kc = 0; kc < KCN; ++kc)
                c = __builtin_amdgcn_wmma_f32_16x16x32_bf16(
                        false, ax[kc], false,
                        load_frag(WihF + (size_t)(nt * 8 + kc) * 512, lane),
                        (short)0, c, false, false);
            const int col = nt * 16 + ln;
#pragma unroll
            for (int rp = 0; rp < 4; ++rp) {
                bf16x2 pr;
                pr[0] = (__bf16)c[2 * rp];
                pr[1] = (__bf16)c[2 * rp + 1];
                *(bf16x2*)(ti + col * 16 + 8 * g + 2 * rp) = pr;
            }
        }
        const int ts1 = dir ? (TT - 2) : 1;
        async_load_x(x + (size_t)ts1 * (BATCH * 256) + bg * (16 * 256),
                     xb + 4096, tid);
        wait_async();
    }
    __syncthreads();

    for (int t = 0; t < TT; ++t) {
        // Defeat LICM: force the weight fragments to be re-loaded from
        // (L2-resident) global memory each step instead of hoisted + spilled.
        {
            unsigned long long p0 = (unsigned long long)WihF;
            unsigned long long p1 = (unsigned long long)WhhF;
            unsigned long long p2 = (unsigned long long)WzF;
            asm volatile("" : "+s"(p0), "+s"(p1), "+s"(p2));
            WihF = (const __bf16*)p0;
            WhhF = (const __bf16*)p1;
            WzF  = (const __bf16*)p2;
        }
        const __bf16* tir = ti + (t & 1) * 16384;   // ti(t), column-major

        // ---- Phase B: th = y @ Whh^T (wave owns n-tiles {c*16+w}) ----
        bf16x16 ay[KCN];
#pragma unroll
        for (int kc = 0; kc < KCN; ++kc)
            ay[kc] = load_a_lds(y_lds, lane, kc * 32);
        f32x8 hb[3];
#pragma unroll
        for (int c3 = 0; c3 < 3; ++c3) {
            const int nt = c3 * 16 + w;
            f32x8 acc = {};
#pragma unroll
            for (int kc = 0; kc < KCN; ++kc)
                acc = __builtin_amdgcn_wmma_f32_16x16x32_bf16(
                        false, ay[kc], false,
                        load_frag(WhhF + (size_t)(nt * 8 + kc) * 512, lane),
                        (short)0, acc, false, false);
            hb[c3] = acc;
        }
        {
            const int j = w * 16 + ln;
#pragma unroll
            for (int r = 0; r < 8; ++r) {
                const int b = r + 8 * g;
                float a1 = (float)tir[(j)*16 + b]        + bias[j];
                float h1 = hb[0][r] + bias[1024 + j];
                float a2 = (float)tir[(256 + j) * 16 + b] + bias[256 + j];
                float h2 = hb[1][r] + bias[1024 + 256 + j];
                float aY = (float)tir[(768 + j) * 16 + b] + bias[768 + j];
                float hY = hb[2][r] + bias[1024 + 512 + j];
                float mb = sigm(a1 + h1);          // DT = 1.0
                float ms = sigm(a2 + h2);
                float zn = (1.f - ms) * zreg[r] + ms * tanhf(aY + hY);
                zreg[r] = zn;
                msb[r]  = mb;
                z_lds[b * 256 + j] = (__bf16)zn;
            }
        }
        // Stage x(t+2) into the buffer phase A finished with last iteration.
        if (t + 2 < TT) {
            const int ts2 = dir ? (TT - 3 - t) : (t + 2);
            async_load_x(x + (size_t)ts2 * (BATCH * 256) + bg * (16 * 256),
                         xb + (t & 1) * 4096, tid);
        }
        __syncthreads();

        // ---- Phase C: u = z @ Wz^T (wave owns n-tile w) + y update ----
        bf16x16 az[KCN];
#pragma unroll
        for (int kc = 0; kc < KCN; ++kc)
            az[kc] = load_a_lds(z_lds, lane, kc * 32);
        {
            f32x8 acc = {};
#pragma unroll
            for (int kc = 0; kc < KCN; ++kc)
                acc = __builtin_amdgcn_wmma_f32_16x16x32_bf16(
                        false, az[kc], false,
                        load_frag(WzF + (size_t)(w * 8 + kc) * 512, lane),
                        (short)0, acc, false, false);
            const int j = w * 16 + ln;
#pragma unroll
            for (int r = 0; r < 8; ++r) {
                const int b = r + 8 * g;
                float aZ = (float)tir[(512 + j) * 16 + b] + bias[512 + j];
                float u  = acc[r] + bias[1792 + j] + aZ;
                float mb = msb[r];
                float yn = (1.f - mb) * yreg[r] + mb * tanhf(u);
                yreg[r] = yn;
                y_lds[b * 256 + j] = (__bf16)yn;
                out[(size_t)t * (BATCH * 512) + (size_t)(bg * 16 + b) * 512 +
                    dir * 256 + j] = yn;
            }
        }

        // ---- Phase A (pipelined): ti(t+1) from the LDS-staged x slab ----
        if (t + 1 < TT) {
            const float* xs = xb + ((t + 1) & 1) * 4096;
            __bf16* tiw = ti + ((t + 1) & 1) * 16384;
            bf16x16 ax[KCN];
#pragma unroll
            for (int kc = 0; kc < KCN; ++kc)
                ax[kc] = load_a_f32(xs, lane, kc * 32);
#pragma unroll
            for (int i = 0; i < 4; ++i) {
                const int nt = w * 4 + i;
                f32x8 c = {};
#pragma unroll
                for (int kc = 0; kc < KCN; ++kc)
                    c = __builtin_amdgcn_wmma_f32_16x16x32_bf16(
                            false, ax[kc], false,
                            load_frag(WihF + (size_t)(nt * 8 + kc) * 512, lane),
                            (short)0, c, false, false);
                const int col = nt * 16 + ln;
#pragma unroll
                for (int rp = 0; rp < 4; ++rp) {
                    bf16x2 pr;
                    pr[0] = (__bf16)c[2 * rp];
                    pr[1] = (__bf16)c[2 * rp + 1];
                    *(bf16x2*)(tiw + col * 16 + 8 * g + 2 * rp) = pr;
                }
            }
        }
        wait_async();     // x(t+2) slab landed in LDS (own wave's ASYNCcnt)
        __syncthreads();  // ...and everyone's; ti(t+1)/y visible
    }
}

// ---------------------------------------------------------------------------
extern "C" void kernel_launch(void* const* d_in, const int* in_sizes, int n_in,
                              void* d_out, int out_size, void* d_ws,
                              size_t ws_size, hipStream_t stream) {
    (void)in_sizes; (void)n_in; (void)out_size; (void)ws_size;
    const float* x    = (const float*)d_in[0];
    const float* fWih = (const float*)d_in[1];
    const float* fbih = (const float*)d_in[2];
    const float* fWhh = (const float*)d_in[3];
    const float* fbhh = (const float*)d_in[4];
    const float* fWz  = (const float*)d_in[5];
    const float* fbz  = (const float*)d_in[6];
    const float* bWih = (const float*)d_in[7];
    const float* bbih = (const float*)d_in[8];
    const float* bWhh = (const float*)d_in[9];
    const float* bbhh = (const float*)d_in[10];
    const float* bWz  = (const float*)d_in[11];
    const float* bbz  = (const float*)d_in[12];

    __bf16* wsb = (__bf16*)d_ws;   // needs 2 MB workspace (bf16 weights)

    swizzle_w<<<64 * 8, 256, 0, stream>>>(fWih, wsb);
    swizzle_w<<<48 * 8, 256, 0, stream>>>(fWhh, wsb + WIH_ELEMS);
    swizzle_w<<<16 * 8, 256, 0, stream>>>(fWz,  wsb + WIH_ELEMS + WHH_ELEMS);
    swizzle_w<<<64 * 8, 256, 0, stream>>>(bWih, wsb + DIR_STRIDE);
    swizzle_w<<<48 * 8, 256, 0, stream>>>(bWhh, wsb + DIR_STRIDE + WIH_ELEMS);
    swizzle_w<<<16 * 8, 256, 0, stream>>>(bWz,
                                          wsb + DIR_STRIDE + WIH_ELEMS + WHH_ELEMS);

    lem_scan<<<8, 512, SMEM_BYTES, stream>>>(x, wsb, fbih, fbhh, fbz,
                                             bbih, bbhh, bbz, (float*)d_out);
}
